// SiglipAttention_82076825027357
// MI455X (gfx1250) — matrix operand order repaired and checked by reading, not compiled, not run on hardware
//
#include <hip/hip_runtime.h>
#include <hip/hip_bf16.h>
#include <math.h>

#define B_   32
#define S_   729
#define D_   1152
#define H_   16
#define HD_  72
#define HDP_ 96
#define SP_  736
#define BH_  (B_ * H_)
#define M_   (B_ * S_)   // 23328

typedef __attribute__((ext_vector_type(8)))  _Float16 v8h;
typedef __attribute__((ext_vector_type(16))) _Float16 v16h;
typedef __attribute__((ext_vector_type(8)))  float    v8f;

__device__ __forceinline__ v8f wmma16x16x32(v16h a, v16h b, v8f c) {
  // D = A(16x32 f16) x B(32x16 f16) + C(16x16 f32)
  return __builtin_amdgcn_wmma_f32_16x16x32_f16(false, a, false, b, (short)0, c,
                                                false, false);
}

__device__ __forceinline__ v8f zero8f() {
  v8f z;
#pragma unroll
  for (int e = 0; e < 8; ++e) z[e] = 0.f;
  return z;
}

// A fragment (16x32, f16 src). Row index pre-clamped by caller: out-of-range
// rows produce garbage C rows that the epilogue never stores.
// ISA 16-bit A layout: lane half h: elems 0..7 = K kb+8h..+7, elems 8..15 = kb+16+8h..+7
__device__ __forceinline__ v16h load_a_f16(const _Float16* __restrict__ src, int ld,
                                           int row, int kb, int h) {
  const _Float16* p = src + (size_t)row * ld + kb + h * 8;
  v8h c0 = *(const v8h*)(p);
  v8h c1 = *(const v8h*)(p + 16);
  v16h f;
#pragma unroll
  for (int e = 0; e < 8; ++e) { f[e] = c0[e]; f[8 + e] = c1[e]; }
  return f;
}

// B fragment (32x16) for A@B^T with B stored row-major [N, K]. Column index
// pre-clamped by caller (garbage C columns discarded at store).
// ISA 16-bit B layout: lane = column N, half h: elems 0..7 = K kb+8h..+7, elems 8..15 = kb+16+8h..+7
__device__ __forceinline__ v16h load_b_f16(const _Float16* __restrict__ src, int ld,
                                           int row, int kb, int h) {
  const _Float16* p = src + (size_t)row * ld + kb + h * 8;
  v8h c0 = *(const v8h*)(p);
  v8h c1 = *(const v8h*)(p + 16);
  v16h f;
#pragma unroll
  for (int e = 0; e < 8; ++e) { f[e] = c0[e]; f[8 + e] = c1[e]; }
  return f;
}

// A fragment from f32 source, unguarded (caller guarantees k in range).
__device__ __forceinline__ v16h load_a_f32_full(const float* __restrict__ p, int kb,
                                                int h) {
  v16h f;
#pragma unroll
  for (int e = 0; e < 8; ++e) {
    f[e]     = (_Float16)p[kb + h * 8 + e];
    f[8 + e] = (_Float16)p[kb + 16 + h * 8 + e];
  }
  return f;
}

// A fragment from f32 source with contraction-dim guard (tail tile only):
// columns >= k_lim contribute zeros.
__device__ __forceinline__ v16h load_a_f32_tail(const float* __restrict__ p, int kb,
                                                int k_lim, int h) {
  v16h f;
#pragma unroll
  for (int e = 0; e < 8; ++e) {
    const int k0 = kb + h * 8 + e;
    const int k1 = kb + 16 + h * 8 + e;
    f[e]     = (k0 < k_lim) ? (_Float16)p[k0] : (_Float16)0.f;
    f[8 + e] = (k1 < k_lim) ? (_Float16)p[k1] : (_Float16)0.f;
  }
  return f;
}

// ---------------------------------------------------------------------------
// Elementwise helpers
// ---------------------------------------------------------------------------
__global__ __launch_bounds__(256) void cvt_f32_f16_kernel(const float* __restrict__ src,
                                                          _Float16* __restrict__ dst,
                                                          long n) {
  const long i = (long)blockIdx.x * 256 + threadIdx.x;
  if (i < n) dst[i] = (_Float16)src[i];
}

__global__ __launch_bounds__(256) void zero_f16_kernel(_Float16* __restrict__ dst,
                                                       long n8) {
  const long i = (long)blockIdx.x * 256 + threadIdx.x;
  if (i < n8) {
    v8h z;
#pragma unroll
    for (int e = 0; e < 8; ++e) z[e] = (_Float16)0.f;
    ((v8h*)dst)[i] = z;
  }
}

// ---------------------------------------------------------------------------
// Projection GEMM: C[M,N] = A[M,K] @ W[N,K]^T + bias
// mode 0: f16 -> q/k padded layout [BH, SP, HDP]
// mode 1: f16 -> v transposed layout [BH, HDP, SP]
// mode 2: f32 -> plain [M, N] (final out projection)
// Block = 256 threads (8 waves), block tile 128(M) x 128(N), wave tile 32x64.
// ---------------------------------------------------------------------------
__global__ __launch_bounds__(256) void gemm_xw_kernel(
    const _Float16* __restrict__ A, const _Float16* __restrict__ W,
    const float* __restrict__ bias, _Float16* __restrict__ outh,
    float* __restrict__ outf, int M, int N, int K, int mode) {
  const int lane = threadIdx.x & 31, wid = threadIdx.x >> 5;
  const int r = lane & 15, h = lane >> 4;
  const int m0 = blockIdx.y * 128 + (wid >> 1) * 32;
  const int n0 = blockIdx.x * 128 + (wid & 1) * 64;

  // Clamped operand row/column indices (stores are guarded in the epilogue).
  const int ra0 = min(m0 + r, M - 1);
  const int ra1 = min(m0 + 16 + r, M - 1);
  const int rb0 = min(n0 + r, N - 1);
  const int rb1 = min(n0 + 16 + r, N - 1);
  const int rb2 = min(n0 + 32 + r, N - 1);
  const int rb3 = min(n0 + 48 + r, N - 1);

  v8f c[2][4];
#pragma unroll
  for (int i = 0; i < 2; ++i)
#pragma unroll
    for (int j = 0; j < 4; ++j) c[i][j] = zero8f();

  for (int kb = 0; kb < K; kb += 32) {
    __builtin_prefetch(A + (size_t)ra0 * K + min(kb + 32, K - 32), 0, 1);
    v16h a0 = load_a_f16(A, K, ra0, kb, h);
    v16h a1 = load_a_f16(A, K, ra1, kb, h);
    v16h b0 = load_b_f16(W, K, rb0, kb, h);
    v16h b1 = load_b_f16(W, K, rb1, kb, h);
    v16h b2 = load_b_f16(W, K, rb2, kb, h);
    v16h b3 = load_b_f16(W, K, rb3, kb, h);
    c[0][0] = wmma16x16x32(a0, b0, c[0][0]);
    c[0][1] = wmma16x16x32(a0, b1, c[0][1]);
    c[0][2] = wmma16x16x32(a0, b2, c[0][2]);
    c[0][3] = wmma16x16x32(a0, b3, c[0][3]);
    c[1][0] = wmma16x16x32(a1, b0, c[1][0]);
    c[1][1] = wmma16x16x32(a1, b1, c[1][1]);
    c[1][2] = wmma16x16x32(a1, b2, c[1][2]);
    c[1][3] = wmma16x16x32(a1, b3, c[1][3]);
  }

#pragma unroll
  for (int i = 0; i < 2; ++i)
#pragma unroll
    for (int j = 0; j < 4; ++j) {
      const int col = n0 + j * 16 + r;
      if (col >= N) continue;
      const float bv = bias[col];
#pragma unroll
      for (int vv = 0; vv < 8; ++vv) {
        const int row = m0 + i * 16 + vv + 8 * h;
        if (row >= M) continue;
        const float val = c[i][j][vv] + bv;
        if (mode == 2) {
          outf[(size_t)row * N + col] = val;
        } else {
          const int b = row / S_, s = row - b * S_;
          const int hh = col / HD_, hd = col - hh * HD_;
          const size_t idx =
              (mode == 0) ? (((size_t)(b * H_ + hh) * SP_ + s) * HDP_ + hd)
                          : (((size_t)(b * H_ + hh) * HDP_ + hd) * SP_ + s);
          outh[idx] = (_Float16)val;
        }
      }
    }
}

// ---------------------------------------------------------------------------
// Scores: attn[bh, q, k] = SCALE * sum_d Q[bh,q,d] * K[bh,k,d]   (K-dim = 96)
// grid (6, 6, 512); block tile 128(q) x 128(k), wave tile 32x64.
// Q/K buffers are zero-padded to [SP=736, HDP=96], so clamped loads are safe
// and padded lanes contribute zeros.
// ---------------------------------------------------------------------------
__global__ __launch_bounds__(256) void scores_kernel(const _Float16* __restrict__ Q,
                                                     const _Float16* __restrict__ Kb,
                                                     float* __restrict__ attn) {
  constexpr float SCALE = 0.11785113019775793f;  // 72^-0.5
  const int bh = blockIdx.z;
  const _Float16* q = Q + (size_t)bh * SP_ * HDP_;
  const _Float16* k = Kb + (size_t)bh * SP_ * HDP_;
  const int lane = threadIdx.x & 31, wid = threadIdx.x >> 5;
  const int r = lane & 15, h = lane >> 4;
  const int m0 = blockIdx.y * 128 + (wid >> 1) * 32;
  const int n0 = blockIdx.x * 128 + (wid & 1) * 64;

  const int ra0 = min(m0 + r, SP_ - 1);
  const int ra1 = min(m0 + 16 + r, SP_ - 1);
  const int rb0 = min(n0 + r, SP_ - 1);
  const int rb1 = min(n0 + 16 + r, SP_ - 1);
  const int rb2 = min(n0 + 32 + r, SP_ - 1);
  const int rb3 = min(n0 + 48 + r, SP_ - 1);

  v8f c[2][4];
#pragma unroll
  for (int i = 0; i < 2; ++i)
#pragma unroll
    for (int j = 0; j < 4; ++j) c[i][j] = zero8f();

#pragma unroll
  for (int kb = 0; kb < HDP_; kb += 32) {
    v16h a0 = load_a_f16(q, HDP_, ra0, kb, h);
    v16h a1 = load_a_f16(q, HDP_, ra1, kb, h);
    v16h b0 = load_b_f16(k, HDP_, rb0, kb, h);
    v16h b1 = load_b_f16(k, HDP_, rb1, kb, h);
    v16h b2 = load_b_f16(k, HDP_, rb2, kb, h);
    v16h b3 = load_b_f16(k, HDP_, rb3, kb, h);
    c[0][0] = wmma16x16x32(a0, b0, c[0][0]);
    c[0][1] = wmma16x16x32(a0, b1, c[0][1]);
    c[0][2] = wmma16x16x32(a0, b2, c[0][2]);
    c[0][3] = wmma16x16x32(a0, b3, c[0][3]);
    c[1][0] = wmma16x16x32(a1, b0, c[1][0]);
    c[1][1] = wmma16x16x32(a1, b1, c[1][1]);
    c[1][2] = wmma16x16x32(a1, b2, c[1][2]);
    c[1][3] = wmma16x16x32(a1, b3, c[1][3]);
  }

  float* outp = attn + (size_t)bh * S_ * S_;
#pragma unroll
  for (int i = 0; i < 2; ++i)
#pragma unroll
    for (int j = 0; j < 4; ++j) {
      const int col = n0 + j * 16 + r;
      if (col >= S_) continue;
#pragma unroll
      for (int vv = 0; vv < 8; ++vv) {
        const int row = m0 + i * 16 + vv + 8 * h;
        if (row >= S_) continue;
        outp[(size_t)row * S_ + col] = c[i][j][vv] * SCALE;
      }
    }
}

// ---------------------------------------------------------------------------
// In-place row softmax over 729 fp32 values. grid (729, 512), block 256.
// ---------------------------------------------------------------------------
__global__ __launch_bounds__(256) void softmax_kernel(float* __restrict__ attn) {
  float* p = attn + ((size_t)blockIdx.y * S_ + blockIdx.x) * S_;
  __shared__ float sm[8];
  const int tid = threadIdx.x;

  float vals[3];
  float m = -3.402823466e38f;
#pragma unroll
  for (int t = 0; t < 3; ++t) {
    const int i = tid + t * 256;
    vals[t] = (i < S_) ? p[i] : -3.402823466e38f;
    m = fmaxf(m, vals[t]);
  }
#pragma unroll
  for (int o = 16; o > 0; o >>= 1) m = fmaxf(m, __shfl_xor(m, o, 32));
  if ((tid & 31) == 0) sm[tid >> 5] = m;
  __syncthreads();
  m = sm[0];
#pragma unroll
  for (int w = 1; w < 8; ++w) m = fmaxf(m, sm[w]);
  __syncthreads();

  float s = 0.f;
#pragma unroll
  for (int t = 0; t < 3; ++t) {
    const int i = tid + t * 256;
    vals[t] = (i < S_) ? __expf(vals[t] - m) : 0.f;
    s += vals[t];
  }
#pragma unroll
  for (int o = 16; o > 0; o >>= 1) s += __shfl_xor(s, o, 32);
  if ((tid & 31) == 0) sm[tid >> 5] = s;
  __syncthreads();
  s = sm[0];
#pragma unroll
  for (int w = 1; w < 8; ++w) s += sm[w];
  const float inv = 1.f / s;

#pragma unroll
  for (int t = 0; t < 3; ++t) {
    const int i = tid + t * 256;
    if (i < S_) p[i] = vals[t] * inv;
  }
}

// ---------------------------------------------------------------------------
// PV: ctx[b*S+s, h*72+hd] = sum_k attn[bh,s,k] * V[bh,k,hd]
// attn fp32 converted on load; V stored transposed [BH, HDP, SP] f16 with
// zero pad for k in [729, 736). k-loop: 22 unguarded tiles + guarded tail.
// grid (1, 6, 512); block tile 128(q) x 96(hd), wave tile 32x48.
// ---------------------------------------------------------------------------
__global__ __launch_bounds__(256) void pv_kernel(const float* __restrict__ attn,
                                                 const _Float16* __restrict__ vT,
                                                 _Float16* __restrict__ ctx) {
  const int bh = blockIdx.z;
  const float* a = attn + (size_t)bh * S_ * S_;
  const _Float16* v = vT + (size_t)bh * HDP_ * SP_;
  const int lane = threadIdx.x & 31, wid = threadIdx.x >> 5;
  const int r = lane & 15, h = lane >> 4;
  const int m0 = blockIdx.y * 128 + (wid >> 1) * 32;
  const int n0 = (wid & 1) * 48;

  const float* pa0 = a + (size_t)min(m0 + r, S_ - 1) * S_;
  const float* pa1 = a + (size_t)min(m0 + 16 + r, S_ - 1) * S_;
  const int rb0 = n0 + r, rb1 = n0 + 16 + r, rb2 = n0 + 32 + r;  // < 96 always

  v8f c[2][3];
#pragma unroll
  for (int i = 0; i < 2; ++i)
#pragma unroll
    for (int j = 0; j < 3; ++j) c[i][j] = zero8f();

  // 22 full tiles: k in [0, 704), all columns < 729 -> unguarded.
  for (int kb = 0; kb < 704; kb += 32) {
    v16h af[2];
    af[0] = load_a_f32_full(pa0, kb, h);
    af[1] = load_a_f32_full(pa1, kb, h);
    v16h b0 = load_b_f16(v, SP_, rb0, kb, h);
    v16h b1 = load_b_f16(v, SP_, rb1, kb, h);
    v16h b2 = load_b_f16(v, SP_, rb2, kb, h);
#pragma unroll
    for (int i = 0; i < 2; ++i) {
      c[i][0] = wmma16x16x32(af[i], b0, c[i][0]);
      c[i][1] = wmma16x16x32(af[i], b1, c[i][1]);
      c[i][2] = wmma16x16x32(af[i], b2, c[i][2]);
    }
  }
  // Tail tile: k in [704, 736), columns >= 729 forced to zero.
  {
    const int kb = 704;
    v16h af[2];
    af[0] = load_a_f32_tail(pa0, kb, S_, h);
    af[1] = load_a_f32_tail(pa1, kb, S_, h);
    v16h b0 = load_b_f16(v, SP_, rb0, kb, h);
    v16h b1 = load_b_f16(v, SP_, rb1, kb, h);
    v16h b2 = load_b_f16(v, SP_, rb2, kb, h);
#pragma unroll
    for (int i = 0; i < 2; ++i) {
      c[i][0] = wmma16x16x32(af[i], b0, c[i][0]);
      c[i][1] = wmma16x16x32(af[i], b1, c[i][1]);
      c[i][2] = wmma16x16x32(af[i], b2, c[i][2]);
    }
  }

  const int b = bh >> 4, hh = bh & 15;
#pragma unroll
  for (int i = 0; i < 2; ++i)
#pragma unroll
    for (int j = 0; j < 3; ++j) {
      const int col = n0 + j * 16 + r;  // hd in [0, 96)
      if (col >= HD_) continue;
#pragma unroll
      for (int vv = 0; vv < 8; ++vv) {
        const int row = m0 + i * 16 + vv + 8 * h;  // s
        if (row >= S_) continue;
        ctx[((size_t)(b * S_ + row)) * D_ + hh * HD_ + col] = (_Float16)c[i][j][vv];
      }
    }
}

// ---------------------------------------------------------------------------
extern "C" void kernel_launch(void* const* d_in, const int* in_sizes, int n_in,
                              void* d_out, int out_size, void* d_ws, size_t ws_size,
                              hipStream_t stream) {
  (void)in_sizes; (void)n_in; (void)out_size; (void)ws_size;

  const float* x   = (const float*)d_in[0];
  const float* q_w = (const float*)d_in[1];
  const float* q_b = (const float*)d_in[2];
  const float* k_w = (const float*)d_in[3];
  const float* k_b = (const float*)d_in[4];
  const float* v_w = (const float*)d_in[5];
  const float* v_b = (const float*)d_in[6];
  const float* o_w = (const float*)d_in[7];
  const float* o_b = (const float*)d_in[8];

  float* out0 = (float*)d_out;                       // [B, S, D]
  float* attn = out0 + (size_t)B_ * S_ * D_;         // [B*H, S, S]

  // Workspace layout (all offsets 256B-aligned)
  char* ws = (char*)d_ws;
  const size_t SZ_XH  = (size_t)M_ * D_ * 2;         // 53,747,712
  const size_t SZ_W   = (size_t)D_ * D_ * 2;         //  2,654,208
  const size_t SZ_QKV = (size_t)BH_ * SP_ * HDP_ * 2;// 72,351,744
  _Float16* xh  = (_Float16*)(ws);
  _Float16* qwh = (_Float16*)(ws + SZ_XH);
  _Float16* kwh = (_Float16*)(ws + SZ_XH + SZ_W);
  _Float16* vwh = (_Float16*)(ws + SZ_XH + 2 * SZ_W);
  _Float16* owh = (_Float16*)(ws + SZ_XH + 3 * SZ_W);
  _Float16* qb  = (_Float16*)(ws + SZ_XH + 4 * SZ_W);
  _Float16* kb  = (_Float16*)(ws + SZ_XH + 4 * SZ_W + SZ_QKV);
  _Float16* vT  = (_Float16*)(ws + SZ_XH + 4 * SZ_W + 2 * SZ_QKV);
  _Float16* ctx = (_Float16*)(ws + SZ_XH + 4 * SZ_W + 3 * SZ_QKV);

  // 1) fp32 -> f16 conversions
  {
    const long nx = (long)M_ * D_;
    cvt_f32_f16_kernel<<<dim3((nx + 255) / 256), dim3(256), 0, stream>>>(x, xh, nx);
    const long nw = (long)D_ * D_;
    const dim3 gw((nw + 255) / 256);
    cvt_f32_f16_kernel<<<gw, dim3(256), 0, stream>>>(q_w, qwh, nw);
    cvt_f32_f16_kernel<<<gw, dim3(256), 0, stream>>>(k_w, kwh, nw);
    cvt_f32_f16_kernel<<<gw, dim3(256), 0, stream>>>(v_w, vwh, nw);
    cvt_f32_f16_kernel<<<gw, dim3(256), 0, stream>>>(o_w, owh, nw);
  }

  // 2) zero padded q/k/vT buffers (contiguous: 3 * SZ_QKV bytes)
  {
    const long n8 = (long)(3 * SZ_QKV) / 16;  // v8h elements
    zero_f16_kernel<<<dim3((n8 + 255) / 256), dim3(256), 0, stream>>>(qb, n8);
  }

  // 3) Q/K/V projections (block tile 128x128)
  {
    const dim3 g(D_ / 128, (M_ + 127) / 128);  // (9, 183)
    gemm_xw_kernel<<<g, dim3(256), 0, stream>>>(xh, qwh, q_b, qb, nullptr,
                                                M_, D_, D_, 0);
    gemm_xw_kernel<<<g, dim3(256), 0, stream>>>(xh, kwh, k_b, kb, nullptr,
                                                M_, D_, D_, 0);
    gemm_xw_kernel<<<g, dim3(256), 0, stream>>>(xh, vwh, v_b, vT, nullptr,
                                                M_, D_, D_, 1);
  }

  // 4) scores = scale * Q K^T  -> attn (fp32, pre-softmax)
  scores_kernel<<<dim3(6, 6, BH_), dim3(256), 0, stream>>>(qb, kb, attn);

  // 5) softmax rows in place
  softmax_kernel<<<dim3(S_, BH_), dim3(256), 0, stream>>>(attn);

  // 6) ctx = attn @ V
  pv_kernel<<<dim3(1, 6, BH_), dim3(256), 0, stream>>>(attn, vT, ctx);

  // 7) out = ctx @ o_w^T + o_b
  {
    const dim3 g(D_ / 128, (M_ + 127) / 128);  // (9, 183)
    gemm_xw_kernel<<<g, dim3(256), 0, stream>>>(ctx, owh, o_b, nullptr, out0,
                                                M_, D_, D_, 2);
  }
}
